// ModelNoteAttention_76905684402774
// MI455X (gfx1250) — matrix operand order repaired and checked by reading, not compiled, not run on hardware
//
#include <hip/hip_runtime.h>
#include <hip/hip_bf16.h>

// ---------------------------------------------------------------------------
// ModelNoteAttention forward for MI455X (gfx1250, wave32, WMMA).
// All K>=32 matmuls run through v_wmma_f32_16x16x32_f16 (f16 in, f32 acc).
// Each GEMM wave computes a 16x(NT*16) strip so the A fragment is reused
// across NT B-tiles (2-4x less A traffic). global_prefetch_b8 pulls the next
// K-slab while WMMAs drain. T=16 scan unrolled as a host launch sequence.
// ---------------------------------------------------------------------------

#define Cc   512
#define CSd  48
#define GSd  32
#define HID  256
#define TAT  32
#define NOTEd 768
#define NHd  8
#define NATTd 64
#define Bb   8
#define Tt   16
#define OUTd 512
#define NEGv (-1000000000.0f)
#define EPSv 1e-5f

typedef __attribute__((ext_vector_type(16))) _Float16 v16h;
typedef __attribute__((ext_vector_type(8)))  float    v8f;

// --- WMMA operand loaders (CDNA5 wave32 layouts, CDNA5 ISA) ------
// A 16x32 f16: lane = row (lane&15); halves split K in groups of 8:
//   elem i (0..7):  k = k0 + half*8 + i
//   elem i (8..15): k = k0 + 16 + half*8 + (i-8)
__device__ inline v16h load_a16(const _Float16* __restrict__ A, int lda,
                                int row, int k0, int half) {
  v16h a;
  const _Float16* p = A + (size_t)row * lda + k0 + half * 8;
#pragma unroll
  for (int i = 0; i < 8; ++i) a[i] = p[i];
#pragma unroll
  for (int i = 0; i < 8; ++i) a[8 + i] = p[16 + i];
  return a;
}
// B 32x16 f16 supplied as BT (N x K row-major): lane = col (lane&15);
//   elem i (0..7): k = k0 + half*8 + i; elem i (8..15): k = k0 + 16 + half*8 + (i-8)
__device__ inline v16h load_bt16(const _Float16* __restrict__ BT, int ldbt,
                                 int col, int k0, int half) {
  v16h b;
  const _Float16* p = BT + (size_t)col * ldbt + k0 + half * 8;
#pragma unroll
  for (int i = 0; i < 8; ++i) b[i] = p[i];
#pragma unroll
  for (int i = 0; i < 8; ++i) b[8 + i] = p[16 + i];
  return b;
}

// --- Generic single-wave WMMA GEMM: D = act(A @ B + bias) -------------------
// Each wave produces a 16x(NT*16) strip (A fragment reused NT times).
// flags: 1=leaky relu, 2=write f32 D, 4=write f16 Dh row-major,
//        8=write f16 Dh transposed (Dh[n*lddh+m])
template <int NT>
__global__ __launch_bounds__(32) void wmma_gemm(
    const _Float16* __restrict__ A, long long aB, int lda,
    const _Float16* __restrict__ BT, long long bB, int ldbt,
    float* __restrict__ D, long long dB, int ldd,
    _Float16* __restrict__ Dh, long long dhB, int lddh,
    const float* __restrict__ bias, const float* __restrict__ bias2, int nsplit,
    int N, int K, int flags) {
  const int lane = threadIdx.x;
  const int half = lane >> 4;
  const int mr = lane & 15;
  const int groups = (N >> 4) / NT;          // column groups of NT tiles
  const int tm = blockIdx.x / groups;
  const int tg = blockIdx.x % groups;
  const int z = blockIdx.z;
  A += (size_t)z * aB;
  BT += (size_t)z * bB;
  v8f acc[NT];
#pragma unroll
  for (int nt = 0; nt < NT; ++nt) acc[nt] = (v8f){};
  for (int k0 = 0; k0 < K; k0 += 32) {
    v16h a = load_a16(A, lda, tm * 16 + mr, k0, half);
    if (k0 + 32 < K) {  // pull next K-slab toward the WGP (global_prefetch_b8)
      __builtin_prefetch(A + (size_t)(tm * 16 + mr) * lda + k0 + 32, 0, 1);
      __builtin_prefetch(
          BT + (size_t)((tg * NT) * 16 + mr) * ldbt + k0 + 32, 0, 1);
    }
#pragma unroll
    for (int nt = 0; nt < NT; ++nt) {
      v16h b = load_bt16(BT, ldbt, (tg * NT + nt) * 16 + mr, k0, half);
      acc[nt] = __builtin_amdgcn_wmma_f32_16x16x32_f16(false, a, false, b,
                                                       (short)0, acc[nt],
                                                       false, false);
    }
  }
#pragma unroll
  for (int nt = 0; nt < NT; ++nt)
#pragma unroll
    for (int r = 0; r < 8; ++r) {
      const int m = tm * 16 + half * 8 + r;   // D layout: VGPR r -> row r+8*half
      const int n = (tg * NT + nt) * 16 + mr; //           lane&15 -> col
      float v = acc[nt][r];
      if (bias) v += (n < nsplit) ? bias[n] : bias2[n - nsplit];
      if (flags & 1) v = (v >= 0.f) ? v : 0.01f * v;
      if (flags & 2) D[(size_t)z * dB + (size_t)m * ldd + n] = v;
      if (flags & 4) Dh[(size_t)z * dhB + (size_t)m * lddh + n] = (_Float16)v;
      if (flags & 8) Dh[(size_t)z * dhB + (size_t)n * lddh + m] = (_Float16)v;
    }
}

// --- prep kernels -----------------------------------------------------------
__global__ void k_transpose_f16(const float* __restrict__ W,
                                _Float16* __restrict__ WT,
                                int K, int N, int Kpad) {
  int i = blockIdx.x * 256 + threadIdx.x;
  if (i >= N * Kpad) return;
  int n = i / Kpad, k = i % Kpad;
  WT[i] = (k < K) ? (_Float16)W[(size_t)k * N + n] : (_Float16)0.f;
}

__global__ void k_f32_to_f16(const float* __restrict__ src, long long sB,
                             _Float16* __restrict__ dst, long long dB, int n) {
  int b = blockIdx.z;
  int i = blockIdx.x * 256 + threadIdx.x;
  if (i < n) dst[(long long)b * dB + i] = (_Float16)src[(long long)b * sB + i];
}

__global__ void k_prep_wdc(const float* __restrict__ Wd,
                           const float* __restrict__ bd,
                           const float* __restrict__ cv,
                           float* __restrict__ wdc) {
  int i = blockIdx.x * 256 + threadIdx.x;
  if (i < 256) {
    float a = 0.f;
    for (int k = 0; k < 32; ++k) a += Wd[i * 32 + k] * cv[k];
    wdc[i] = a;
  } else if (i == 256) {
    float a = 0.f;
    for (int k = 0; k < 32; ++k) a += bd[k] * cv[k];
    wdc[256] = a;
  }
}

// --- per-step kernels -------------------------------------------------------
// K1: cnT (96x512 f16, B-operand of adj@[ce|ne]) + masks [m1(512),m2(512),m3(512)]
__global__ void k_build_cn(const float* __restrict__ code_x,
                           const float* __restrict__ neighbors,
                           const float* __restrict__ divided,
                           const float* __restrict__ c_emb,
                           const float* __restrict__ n_emb,
                           _Float16* __restrict__ cnT,
                           float* __restrict__ m1m23, int t) {
  const int b = blockIdx.z;
  const int idx = blockIdx.x * 256 + threadIdx.x;   // < 96*512
  const int j = idx / 512, c = idx & 511;
  const size_t base = ((size_t)b * Tt + t) * Cc;
  float v;
  if (j < 48) v = code_x[base + c] * c_emb[c * 48 + j];
  else        v = neighbors[base + c] * n_emb[c * 48 + (j - 48)];
  cnT[(size_t)b * (96 * 512) + j * 512 + c] = (_Float16)v;
  if (idx < 1536) {
    const float* dv = divided + base * 3;
    int sec = idx >> 9, c2 = idx & 511;  // 0:m1 1:m2 2:m3
    m1m23[(size_t)b * 1536 + idx] = dv[c2 * 3 + sec];
  }
}

// K3: X (1024x64 f16, K-padded) — rows 0..511 co input, 512..1023 no input
__global__ void k_build_x(const float* __restrict__ code_x,
                          const float* __restrict__ neighbors,
                          const float* __restrict__ c_emb,
                          const float* __restrict__ n_emb,
                          const float* __restrict__ aceane,
                          _Float16* __restrict__ Xh, int t) {
  const int b = blockIdx.z;
  const int idx = blockIdx.x * 256 + threadIdx.x;   // < 1024*64
  const int m = idx >> 6, k = idx & 63;
  float out = 0.f;
  if (k < 48) {
    const int c = m & 511;
    const size_t base = ((size_t)b * Tt + t) * Cc;
    const float ac = aceane[(size_t)b * (512 * 96) + c * 96 + k];
    const float an = aceane[(size_t)b * (512 * 96) + c * 96 + 48 + k];
    if (m < 512) {
      float ci = code_x[base + c];
      out = ci * c_emb[c * 48 + k] + ci * (ac + an);
    } else {
      float ni = neighbors[base + c];
      out = ni * n_emb[c * 48 + k] + ni * (an + ac);
    }
  }
  Xh[(size_t)b * (1024 * 64) + idx] = (_Float16)out;
}

// K5: layernorm rows 0..511 of con -> co_n_h ; rows 512..1023 -> no_cur (f32)
__global__ void k_ln_no(const float* __restrict__ con,
                        const float* __restrict__ ln_g,
                        const float* __restrict__ ln_b,
                        _Float16* __restrict__ conh,
                        float* __restrict__ noCur) {
  const int g = blockIdx.x * 256 + threadIdx.x;   // < 8*1024
  const int b = g >> 10, row = g & 1023;
  const float* src = con + (size_t)b * (1024 * 32) + row * 32;
  if (row < 512) {
    float mu = 0.f;
#pragma unroll
    for (int k = 0; k < 32; ++k) mu += src[k];
    mu *= (1.f / 32.f);
    float var = 0.f;
#pragma unroll
    for (int k = 0; k < 32; ++k) { float d = src[k] - mu; var += d * d; }
    var *= (1.f / 32.f);
    const float rs = rsqrtf(var + EPSv);
#pragma unroll
    for (int k = 0; k < 32; ++k) {
      float v = (src[k] - mu) * rs * ln_g[k] + ln_b[k];
      conh[(size_t)b * (512 * 32) + row * 32 + k] = (_Float16)v;
    }
  } else {
    float* dst = noCur + (size_t)b * (512 * 32) + (row - 512) * 32;
#pragma unroll
    for (int k = 0; k < 32; ++k) dst[k] = src[k];
  }
}

// K6: q_in_h rows 0..511 = f16(no_prev)  (rows 512..1023 = u_emb, set once)
__global__ void k_build_qin(const float* __restrict__ noPrev,
                            _Float16* __restrict__ qinh) {
  const int g = blockIdx.x * 256 + threadIdx.x;   // < 8*512*32
  const int b = g >> 14, i = g & 16383;
  qinh[(size_t)b * (1024 * 32) + i] = (_Float16)noPrev[(size_t)b * 16384 + i];
}

// K8: masked row-max of attention scores (two-pass softmax, pass 1)
__global__ __launch_bounds__(32) void k_att_rowmax(
    const _Float16* __restrict__ qkh, const float* __restrict__ m1m23,
    float* __restrict__ rowmax) {
  const int lane = threadIdx.x, half = lane >> 4, mr = lane & 15;
  const int tm = blockIdx.x, b = blockIdx.z;
  const _Float16* qh = qkh + (size_t)b * (1024 * 64);
  const _Float16* kh = qh + 32;
  const float* m23 = m1m23 + (size_t)b * 1536 + 512;
  const v16h a = load_a16(qh, 64, tm * 16 + mr, 0, half);
  float mx[8];
#pragma unroll
  for (int r = 0; r < 8; ++r) mx[r] = -3.0e38f;
  for (int ct = 0; ct < 64; ++ct) {
    if (ct + 1 < 64)
      __builtin_prefetch(kh + (size_t)((ct + 1) * 16 + mr) * 64, 0, 1);
    v16h bv = load_bt16(kh, 64, ct * 16 + mr, 0, half);
    v8f z = {};
    v8f s = __builtin_amdgcn_wmma_f32_16x16x32_f16(false, a, false, bv,
                                                   (short)0, z, false, false);
    const float msk = m23[ct * 16 + mr];
#pragma unroll
    for (int r = 0; r < 8; ++r) {
      float e = (msk > 0.f) ? s[r] * 0.17677669529663687f : NEGv;
      mx[r] = fmaxf(mx[r], e);
    }
  }
#pragma unroll
  for (int off = 1; off < 16; off <<= 1)
#pragma unroll
    for (int r = 0; r < 8; ++r)
      mx[r] = fmaxf(mx[r], __shfl_xor(mx[r], off, 32));
  if (mr == 0)
#pragma unroll
    for (int r = 0; r < 8; ++r)
      rowmax[(size_t)b * 1024 + tm * 16 + half * 8 + r] = mx[r];
}

// K9: pass 2 — h_m23 = tanh(softmax(sc) @ V). Two waves split N (0..127/128..255).
__global__ __launch_bounds__(64) void k_att_pv(
    const _Float16* __restrict__ qkh, const _Float16* __restrict__ vTh,
    const float* __restrict__ m1m23, const float* __restrict__ rowmax,
    float* __restrict__ hm23) {
  __shared__ _Float16 P[2][16 * 32];
  const int tid = threadIdx.x, w = tid >> 5, lane = tid & 31;
  const int half = lane >> 4, mr = lane & 15;
  const int tm = blockIdx.x, b = blockIdx.z;
  const _Float16* qh = qkh + (size_t)b * (1024 * 64);
  const _Float16* kh = qh + 32;
  const _Float16* vT = vTh + (size_t)b * (256 * 512);
  const float* m23 = m1m23 + (size_t)b * 1536 + 512;
  const v16h aq = load_a16(qh, 64, tm * 16 + mr, 0, half);
  float rm[8], sum8[8];
  v8f acc[8];
#pragma unroll
  for (int r = 0; r < 8; ++r) {
    rm[r] = rowmax[(size_t)b * 1024 + tm * 16 + half * 8 + r];
    sum8[r] = 0.f;
    acc[r] = (v8f){};
  }
  for (int cp = 0; cp < 32; ++cp) {
#pragma unroll
    for (int sub = 0; sub < 2; ++sub) {
      const int ct = cp * 2 + sub;
      v16h bk = load_bt16(kh, 64, ct * 16 + mr, 0, half);
      v8f z = {};
      v8f s = __builtin_amdgcn_wmma_f32_16x16x32_f16(false, aq, false, bk,
                                                     (short)0, z, false, false);
      const float msk = m23[ct * 16 + mr];
#pragma unroll
      for (int r = 0; r < 8; ++r) {
        float e = (msk > 0.f) ? s[r] * 0.17677669529663687f : NEGv;
        float p = __expf(e - rm[r]);
        sum8[r] += p;
        P[w][(half * 8 + r) * 32 + sub * 16 + mr] = (_Float16)p;
      }
    }
    // reload P tile in A layout (wave-local LDS round-trip)
    v16h ap;
#pragma unroll
    for (int i = 0; i < 8; ++i) ap[i] = P[w][mr * 32 + half * 8 + i];
#pragma unroll
    for (int i = 0; i < 8; ++i) ap[8 + i] = P[w][mr * 32 + 16 + half * 8 + i];
    if (cp + 1 < 32)   // prefetch next V K-slab (global_prefetch_b8)
      __builtin_prefetch(
          vT + (size_t)(w * 128 + mr) * 512 + (((cp + 1) * 32) & 511), 0, 1);
#pragma unroll
    for (int nt = 0; nt < 8; ++nt) {
      const int n0 = w * 128 + nt * 16 + mr;
      v16h bv;
#pragma unroll
      for (int i = 0; i < 16; ++i) {
        int kk = cp * 32 + (i < 8 ? half * 8 + i : 16 + half * 8 + (i - 8));
        bv[i] = vT[(size_t)n0 * 512 + (kk & 511)];   // V rows 512..1023 == 0..511
      }
      acc[nt] = __builtin_amdgcn_wmma_f32_16x16x32_f16(false, ap, false, bv,
                                                       (short)0, acc[nt],
                                                       false, false);
    }
  }
#pragma unroll
  for (int off = 1; off < 16; off <<= 1)
#pragma unroll
    for (int r = 0; r < 8; ++r) sum8[r] += __shfl_xor(sum8[r], off, 32);
#pragma unroll
  for (int nt = 0; nt < 8; ++nt)
#pragma unroll
    for (int r = 0; r < 8; ++r) {
      const int m = tm * 16 + half * 8 + r;
      const int n = w * 128 + nt * 16 + mr;
      hm23[(size_t)b * (1024 * 256) + (size_t)m * 256 + n] =
          tanhf(acc[nt][r] / sum8[r]);
    }
}

// GRU elementwise
__global__ void k_gru(const float* __restrict__ gi, const float* __restrict__ gh,
                      const float* __restrict__ h, float* __restrict__ hm1) {
  const int b = blockIdx.z;
  const int g = blockIdx.x * 256 + threadIdx.x;   // < 512*256
  const int c = g >> 8, j = g & 255;
  const size_t gb = (size_t)b * (512 * 768) + (size_t)c * 768;
  const float ir = gi[gb + j], iz = gi[gb + 256 + j], ia = gi[gb + 512 + j];
  const float hr = gh[gb + j], hz = gh[gb + 256 + j], ha = gh[gb + 512 + j];
  const float r = 1.f / (1.f + __expf(-(ir + hr)));
  const float z = 1.f / (1.f + __expf(-(iz + hz)));
  const float a = tanhf(ia + r * ha);
  const size_t hb = (size_t)b * (512 * 256) + (size_t)c * 256 + j;
  hm1[hb] = (1.f - z) * a + z * h[hb];
}

// combine h_new + masked max-pool -> outs (pre note-attn); h, h_h updated
__global__ void k_combine(const float* __restrict__ hm1,
                          const float* __restrict__ hm23,
                          const float* __restrict__ m1m23,
                          float* __restrict__ h, _Float16* __restrict__ hh,
                          float* __restrict__ outs, float gate, int t) {
  const int b = blockIdx.z, j = threadIdx.x;
  const float* M = m1m23 + (size_t)b * 1536;
  const float* A1 = hm1 + (size_t)b * (512 * 256);
  const float* A23 = hm23 + (size_t)b * (1024 * 256);
  float* H = h + (size_t)b * (512 * 256);
  _Float16* Hh = hh + (size_t)b * (512 * 256);
  float o1 = -3.0e38f, o23 = -3.0e38f, m1s = 0.f, m23s = 0.f;
  for (int c = 0; c < 512; ++c) {
    const float m1v = M[c], m2v = M[512 + c], m3v = M[1024 + c];
    const float a1 = A1[(size_t)c * 256 + j];
    const float a2 = A23[(size_t)c * 256 + j];
    const float a3 = A23[(size_t)(512 + c) * 256 + j];
    const float hn = m1v * a1 + gate * (m2v * a2 + m3v * a3);
    H[(size_t)c * 256 + j] = hn;
    Hh[(size_t)c * 256 + j] = (_Float16)hn;
    o1 = fmaxf(o1, (m1v > 0.f) ? a1 : NEGv);
    o23 = fmaxf(o23, (m2v > 0.f) ? a2 : NEGv);
    o23 = fmaxf(o23, (m3v > 0.f) ? a3 : NEGv);
    m1s += m1v;
    m23s += m2v + m3v;
  }
  float out1 = (m1s > 0.f) ? o1 : NEGv;
  float out23 = (gate * m23s > 0.f) ? o23 : NEGv;
  float out = fmaxf(out1, out23);
  if (out <= 0.5f * NEGv) out = 0.f;
  outs[((size_t)b * Tt + t) * 256 + j] = out;
}

// note attention (tiny GEMMs, VALU) — adds into outs[b][t]
__global__ void k_note(const float* __restrict__ note_emb,
                       const float* __restrict__ Wq_n, const float* __restrict__ bq_n,
                       const float* __restrict__ knvn,
                       const float* __restrict__ Wo_n, const float* __restrict__ bo_n,
                       float* __restrict__ outs, int t) {
  __shared__ float qn[512];
  __shared__ float sAtt[8 * 512];
  __shared__ float ctx[512];
  const int b = blockIdx.z, tid = threadIdx.x;
  const float* note = note_emb + ((size_t)b * Tt + t) * NOTEd;
  const float* KV = knvn + (size_t)b * (512 * 1024);
  for (int o = tid; o < 512; o += 256) {
    float a = bq_n[o];
    for (int i = 0; i < NOTEd; ++i) a += note[i] * Wq_n[(size_t)i * 512 + o];
    qn[o] = a;
  }
  __syncthreads();
  for (int e = tid; e < 4096; e += 256) {
    const int hh = e >> 9, c = e & 511;
    float a = 0.f;
    const float* kr = KV + (size_t)c * 1024 + hh * 64;
    const float* qr = qn + hh * 64;
#pragma unroll 8
    for (int k = 0; k < 64; ++k) a += qr[k] * kr[k];
    sAtt[hh * 512 + c] = a * 0.125f;
  }
  __syncthreads();
  {   // per-head softmax: wave w handles head w
    const int w = tid >> 5, lane = tid & 31;
    float mx = -3.0e38f;
    for (int c = lane; c < 512; c += 32) mx = fmaxf(mx, sAtt[w * 512 + c]);
    for (int off = 16; off >= 1; off >>= 1) mx = fmaxf(mx, __shfl_xor(mx, off, 32));
    float sm = 0.f;
    for (int c = lane; c < 512; c += 32) {
      float e = __expf(sAtt[w * 512 + c] - mx);
      sAtt[w * 512 + c] = e;
      sm += e;
    }
    for (int off = 16; off >= 1; off >>= 1) sm += __shfl_xor(sm, off, 32);
    const float inv = 1.f / sm;
    for (int c = lane; c < 512; c += 32) sAtt[w * 512 + c] *= inv;
  }
  __syncthreads();
  for (int o = tid; o < 512; o += 256) {
    const int hh = o >> 6;
    float a = 0.f;
    for (int c = 0; c < 512; ++c)
      a += sAtt[hh * 512 + c] * KV[(size_t)c * 1024 + 512 + o];
    ctx[o] = a;
  }
  __syncthreads();
  {
    float a = bo_n[tid];
    for (int i = 0; i < 512; ++i) a += ctx[i] * Wo_n[(size_t)i * 256 + tid];
    outs[((size_t)b * Tt + t) * 256 + tid] += a;
  }
}

// temporal softmax pooling
__global__ void k_pool(const float* __restrict__ outs,
                       const float* __restrict__ wdc,
                       const int* __restrict__ lens,
                       float* __restrict__ pooled) {
  __shared__ float sc[16];
  const int b = blockIdx.z, tid = threadIdx.x;
  if (tid < 16) {
    float a = wdc[256];
    const float* o = outs + ((size_t)b * Tt + tid) * 256;
    for (int j = 0; j < 256; ++j) a += o[j] * wdc[j];
    sc[tid] = a;
  }
  __syncthreads();
  if (tid == 0) {
    const int L = lens[b];
    float x[16], mx = -3.0e38f;
    for (int t = 0; t < 16; ++t) { x[t] = (t < L) ? sc[t] : NEGv; mx = fmaxf(mx, x[t]); }
    float s = 0.f;
    for (int t = 0; t < 16; ++t) { x[t] = __expf(x[t] - mx); s += x[t]; }
    for (int t = 0; t < 16; ++t) sc[t] = x[t] / s;
  }
  __syncthreads();
  float p = 0.f;
  for (int t = 0; t < 16; ++t) p += outs[((size_t)b * Tt + t) * 256 + tid] * sc[t];
  pooled[b * 256 + tid] = p;
}

__global__ void k_bn(const float* __restrict__ pooled,
                     const float* __restrict__ bn_g, const float* __restrict__ bn_b,
                     float* __restrict__ hnorm) {
  const int j = threadIdx.x;
  float mu = 0.f;
  for (int b = 0; b < 8; ++b) mu += pooled[b * 256 + j];
  mu *= 0.125f;
  float var = 0.f;
  for (int b = 0; b < 8; ++b) { float d = pooled[b * 256 + j] - mu; var += d * d; }
  var *= 0.125f;
  const float rs = rsqrtf(var + EPSv);
  for (int b = 0; b < 8; ++b)
    hnorm[b * 256 + j] = (pooled[b * 256 + j] - mu) * rs * bn_g[j] + bn_b[j];
}

__global__ void k_final(const float* __restrict__ hnorm,
                        const float* __restrict__ Wc, const float* __restrict__ bc,
                        float* __restrict__ out) {
  const int b = blockIdx.z, j = threadIdx.x;
  for (int o = j; o < 512; o += 256) {
    float a = bc[o];
    for (int i = 0; i < 256; ++i) a += hnorm[b * 256 + i] * Wc[(size_t)i * 512 + o];
    out[(size_t)b * 512 + o] = 1.f / (1.f + __expf(-a));
  }
}

// ---------------------------------------------------------------------------
extern "C" void kernel_launch(void* const* d_in, const int* in_sizes, int n_in,
                              void* d_out, int out_size, void* d_ws, size_t ws_size,
                              hipStream_t stream) {
  const float* code_x   = (const float*)d_in[0];
  const float* divided  = (const float*)d_in[1];
  const float* neighbors= (const float*)d_in[2];
  const float* note_emb = (const float*)d_in[3];
  const float* c_emb    = (const float*)d_in[4];
  const float* n_emb    = (const float*)d_in[5];
  const float* u_emb    = (const float*)d_in[6];
  const float* adj      = (const float*)d_in[7];
  const float* Wg  = (const float*)d_in[8];
  const float* bg  = (const float*)d_in[9];
  const float* ln_g= (const float*)d_in[10];
  const float* ln_b= (const float*)d_in[11];
  const float* Wi  = (const float*)d_in[12];
  const float* Wh  = (const float*)d_in[13];
  const float* bi  = (const float*)d_in[14];
  const float* bh  = (const float*)d_in[15];
  const float* Wq_t= (const float*)d_in[16];
  const float* bq_t= (const float*)d_in[17];
  const float* Wk_t= (const float*)d_in[18];
  const float* bk_t= (const float*)d_in[19];
  const float* Wv_t= (const float*)d_in[20];
  const float* bv_t= (const float*)d_in[21];
  const float* Wq_n= (const float*)d_in[22];
  const float* bq_n= (const float*)d_in[23];
  const float* Wk_n= (const float*)d_in[24];
  const float* bk_n= (const float*)d_in[25];
  const float* Wv_n= (const float*)d_in[26];
  const float* bv_n= (const float*)d_in[27];
  const float* Wo_n= (const float*)d_in[28];
  const float* bo_n= (const float*)d_in[29];
  const float* Wd  = (const float*)d_in[30];
  const float* bd  = (const float*)d_in[31];
  const float* ctxv= (const float*)d_in[32];
  const float* bn_g= (const float*)d_in[33];
  const float* bn_b= (const float*)d_in[34];
  const float* Wc  = (const float*)d_in[35];
  const float* bc  = (const float*)d_in[36];
  const int*   lens= (const int*)d_in[37];

  // --- workspace bump allocator --------------------------------------------
  size_t off = 0;
  auto alloc = [&](size_t bytes) -> void* {
    size_t cur = (off + 255) & ~(size_t)255;
    off = cur + bytes;
    return (void*)((char*)d_ws + cur);
  };
  _Float16* adj_h = (_Float16*)alloc((size_t)512 * 512 * 2);
  _Float16* WgT   = (_Float16*)alloc(32 * 64 * 2);
  _Float16* WiT   = (_Float16*)alloc(768 * 32 * 2);
  _Float16* WhT   = (_Float16*)alloc((size_t)768 * 256 * 2);
  _Float16* WqkT  = (_Float16*)alloc(64 * 32 * 2);
  _Float16* WvtT  = (_Float16*)alloc(256 * 32 * 2);
  _Float16* WkvT  = (_Float16*)alloc((size_t)1024 * 256 * 2);
  float*    wdc   = (float*)alloc(257 * 4);
  _Float16* qinh  = (_Float16*)alloc((size_t)8 * 1024 * 32 * 2);
  _Float16* cnT   = (_Float16*)alloc((size_t)8 * 96 * 512 * 2);
  float*    m1m23 = (float*)alloc((size_t)8 * 1536 * 4);
  float*    aceane= (float*)alloc((size_t)8 * 512 * 96 * 4);
  _Float16* Xh    = (_Float16*)alloc((size_t)8 * 1024 * 64 * 2);
  float*    con   = (float*)alloc((size_t)8 * 1024 * 32 * 4);
  _Float16* conh  = (_Float16*)alloc((size_t)8 * 512 * 32 * 2);
  float*    nobuf = (float*)alloc((size_t)2 * 8 * 512 * 32 * 4);
  _Float16* qkh   = (_Float16*)alloc((size_t)8 * 1024 * 64 * 2);
  _Float16* vTh   = (_Float16*)alloc((size_t)8 * 256 * 512 * 2);
  float*    rowmax= (float*)alloc((size_t)8 * 1024 * 4);
  float*    hm23  = (float*)alloc((size_t)8 * 1024 * 256 * 4);
  float*    gi    = (float*)alloc((size_t)8 * 512 * 768 * 4);
  float*    gh    = (float*)alloc((size_t)8 * 512 * 768 * 4);
  float*    hm1   = (float*)alloc((size_t)8 * 512 * 256 * 4);
  float*    hbuf  = (float*)alloc((size_t)8 * 512 * 256 * 4);
  _Float16* hh    = (_Float16*)alloc((size_t)8 * 512 * 256 * 2);
  float*    knvn  = (float*)alloc((size_t)8 * 512 * 1024 * 4);
  float*    outs  = (float*)alloc((size_t)8 * 16 * 256 * 4);
  float*    pooled= (float*)alloc((size_t)8 * 256 * 4);
  float*    hnorm = (float*)alloc((size_t)8 * 256 * 4);
  (void)ws_size; (void)in_sizes; (void)n_in; (void)out_size;

  const int BIG = 1 << 30;
  const dim3 blk256(256), blk64(64), blk32(32);

  // --- one-time prep (re-done each call: stateless/deterministic) ----------
  k_f32_to_f16<<<dim3((512 * 512 + 255) / 256, 1, 1), blk256, 0, stream>>>(
      adj, 0, adj_h, 0, 512 * 512);
  k_transpose_f16<<<dim3((32 * 64 + 255) / 256), blk256, 0, stream>>>(Wg, WgT, 48, 32, 64);
  k_transpose_f16<<<dim3((768 * 32 + 255) / 256), blk256, 0, stream>>>(Wi, WiT, 32, 768, 32);
  k_transpose_f16<<<dim3((768 * 256 + 255) / 256), blk256, 0, stream>>>(Wh, WhT, 256, 768, 256);
  k_transpose_f16<<<dim3((32 * 32 + 255) / 256), blk256, 0, stream>>>(Wq_t, WqkT, 32, 32, 32);
  k_transpose_f16<<<dim3((32 * 32 + 255) / 256), blk256, 0, stream>>>(Wk_t, WqkT + 32 * 32, 32, 32, 32);
  k_transpose_f16<<<dim3((256 * 32 + 255) / 256), blk256, 0, stream>>>(Wv_t, WvtT, 32, 256, 32);
  k_transpose_f16<<<dim3((512 * 256 + 255) / 256), blk256, 0, stream>>>(Wk_n, WkvT, 256, 512, 256);
  k_transpose_f16<<<dim3((512 * 256 + 255) / 256), blk256, 0, stream>>>(Wv_n, WkvT + (size_t)512 * 256, 256, 512, 256);
  k_prep_wdc<<<dim3(2), blk256, 0, stream>>>(Wd, bd, ctxv, wdc);
  // static part of q_in (rows 512..1023 = u_emb), per patient
  k_f32_to_f16<<<dim3((512 * 32 + 255) / 256, 1, 8), blk256, 0, stream>>>(
      u_emb, 0, qinh + 512 * 32, 1024 * 32, 512 * 32);
  hipMemsetAsync(hbuf, 0, (size_t)8 * 512 * 256 * 4, stream);
  hipMemsetAsync(hh, 0, (size_t)8 * 512 * 256 * 2, stream);
  hipMemsetAsync(nobuf, 0, (size_t)8 * 512 * 32 * 4, stream);   // ping buffer 0

  // --- recurrent scan, unrolled on host ------------------------------------
  for (int t = 0; t < Tt; ++t) {
    const float gate = (t > 0) ? 1.f : 0.f;
    float* noPrev = nobuf + (size_t)(t & 1) * 8 * 512 * 32;
    float* noCur  = nobuf + (size_t)((t + 1) & 1) * 8 * 512 * 32;

    k_build_cn<<<dim3(96 * 512 / 256, 1, 8), blk256, 0, stream>>>(
        code_x, neighbors, divided, c_emb, n_emb, cnT, m1m23, t);
    // [ace|ane] = adj @ [ce|ne] : M=512 N=96 K=512 (NT=2 -> 16x32 strips)
    wmma_gemm<2><<<dim3(32 * 3, 1, 8), blk32, 0, stream>>>(
        adj_h, 0, 512, cnT, (long long)96 * 512, 512,
        aceane, (long long)512 * 96, 96, nullptr, 0, 0,
        nullptr, nullptr, 0, 96, 512, 2);
    k_build_x<<<dim3(1024 * 64 / 256, 1, 8), blk256, 0, stream>>>(
        code_x, neighbors, c_emb, n_emb, aceane, Xh, t);
    // [co|no] = leaky(X @ Wg + bg) : M=1024 N=32 K=64
    wmma_gemm<2><<<dim3(64 * 1, 1, 8), blk32, 0, stream>>>(
        Xh, (long long)1024 * 64, 64, WgT, 0, 64,
        con, (long long)1024 * 32, 32, nullptr, 0, 0,
        bg, bg, BIG, 32, 64, 2 | 1);
    k_ln_no<<<dim3(8 * 1024 / 256), blk256, 0, stream>>>(con, ln_g, ln_b, conh, noCur);
    k_build_qin<<<dim3(8 * 512 * 32 / 256), blk256, 0, stream>>>(noPrev, qinh);
    // [q|k] = q_in @ [Wq_t|Wk_t] + [bq|bk] : M=1024 N=64 K=32 (f16 out)
    wmma_gemm<4><<<dim3(64 * 1, 1, 8), blk32, 0, stream>>>(
        qinh, (long long)1024 * 32, 32, WqkT, 0, 32,
        nullptr, 0, 0, qkh, (long long)1024 * 64, 64,
        bq_t, bk_t, 32, 64, 32, 4);
    // vT = (co_n @ Wv_t + bv_t)^T : M=512 N=256 K=32 (f16 transposed out)
    wmma_gemm<4><<<dim3(32 * 4, 1, 8), blk32, 0, stream>>>(
        conh, (long long)512 * 32, 32, WvtT, 0, 32,
        nullptr, 0, 0, vTh, (long long)256 * 512, 512,
        bv_t, bv_t, BIG, 256, 32, 8);
    k_att_rowmax<<<dim3(64, 1, 8), blk32, 0, stream>>>(qkh, m1m23, rowmax);
    k_att_pv<<<dim3(64, 1, 8), blk64, 0, stream>>>(qkh, vTh, m1m23, rowmax, hm23);
    // gi = co_n @ Wi + bi : M=512 N=768 K=32
    wmma_gemm<4><<<dim3(32 * 12, 1, 8), blk32, 0, stream>>>(
        conh, (long long)512 * 32, 32, WiT, 0, 32,
        gi, (long long)512 * 768, 768, nullptr, 0, 0,
        bi, bi, BIG, 768, 32, 2);
    // gh = h_prev @ Wh + bh : M=512 N=768 K=256
    wmma_gemm<4><<<dim3(32 * 12, 1, 8), blk32, 0, stream>>>(
        hh, (long long)512 * 256, 256, WhT, 0, 256,
        gh, (long long)512 * 768, 768, nullptr, 0, 0,
        bh, bh, BIG, 768, 256, 2);
    k_gru<<<dim3(512 * 256 / 256, 1, 8), blk256, 0, stream>>>(gi, gh, hbuf, hm1);
    k_combine<<<dim3(1, 1, 8), blk256, 0, stream>>>(hm1, hm23, m1m23, hbuf, hh,
                                                    outs, gate, t);
    // [kn|vn] = h_new @ [Wk_n|Wv_n] + [bk_n|bv_n] : M=512 N=1024 K=256
    wmma_gemm<4><<<dim3(32 * 16, 1, 8), blk32, 0, stream>>>(
        hh, (long long)512 * 256, 256, WkvT, 0, 256,
        knvn, (long long)512 * 1024, 1024, nullptr, 0, 0,
        bk_n, bv_n, 512, 1024, 256, 2);
    k_note<<<dim3(1, 1, 8), blk256, 0, stream>>>(note_emb, Wq_n, bq_n, knvn,
                                                 Wo_n, bo_n, outs, t);
  }

  // --- pooling, batchnorm, classifier --------------------------------------
  k_pool<<<dim3(1, 1, 8), blk256, 0, stream>>>(outs, wdc, lens, pooled);
  k_bn<<<dim3(1), blk256, 0, stream>>>(pooled, bn_g, bn_b, hnorm);
  k_final<<<dim3(1, 1, 8), blk256, 0, stream>>>(hnorm, Wc, bc, (float*)d_out);
}
